// HistogramFeatureExtractor_84645215469718
// MI455X (gfx1250) — compile-verified, matrix operand order checked by reading, not run on hardware
//
#include <hip/hip_runtime.h>
#include <hip/hip_bf16.h>

#define BINS 128
#define B_DIM 64
#define C_DIM 3
#define HW (512 * 512)
#define NSLICES (B_DIM * C_DIM)          // 192
#define KTOT (C_DIM * BINS)              // 384
#define NOUT 128
#define BLOCKS_PER_SLICE 64
#define HIST_THREADS 256
#define ELEMS_PER_BLOCK (HW / BLOCKS_PER_SLICE)   // 4096
#define WAVES_PER_BLOCK (HIST_THREADS / 32)       // 8

typedef __attribute__((ext_vector_type(16))) _Float16 v16h;
typedef __attribute__((ext_vector_type(8)))  float    v8f;

// ---------------------------------------------------------------------------
// Pass 0: zero the global count buffer (must be re-done each launch).
// ---------------------------------------------------------------------------
__global__ void hist_zero_kernel(unsigned int* __restrict__ counts, int n) {
    int i = blockIdx.x * blockDim.x + threadIdx.x;
    if (i < n) counts[i] = 0u;
}

// ---------------------------------------------------------------------------
// Pass 1: per-(b,c) 128-bin histogram. HBM-bound: 201 MB read once.
// Vectorized float4 loads (global_load_b128), per-wave LDS histograms
// (ds_add_u32), one global atomicAdd per bin per block to flush.
// ---------------------------------------------------------------------------
__global__ __launch_bounds__(HIST_THREADS)
void hist_kernel(const float* __restrict__ x, unsigned int* __restrict__ counts) {
    __shared__ unsigned int lh[WAVES_PER_BLOCK][BINS];

    const int tid  = threadIdx.x;
    const int wave = tid >> 5;

    // zero LDS histograms
    #pragma unroll
    for (int i = tid; i < WAVES_PER_BLOCK * BINS; i += HIST_THREADS)
        ((unsigned int*)lh)[i] = 0u;
    __syncthreads();

    const int slice = blockIdx.x / BLOCKS_PER_SLICE;
    const int chunk = blockIdx.x % BLOCKS_PER_SLICE;

    const float4* __restrict__ base =
        (const float4*)(x + (size_t)slice * HW + (size_t)chunk * ELEMS_PER_BLOCK);

    // 4096 elements per block = 1024 float4, 4 per thread, coalesced.
    #pragma unroll
    for (int i = 0; i < 4; ++i) {
        float4 v = base[tid + i * HIST_THREADS];
        float c[4] = {v.x, v.y, v.z, v.w};
        #pragma unroll
        for (int j = 0; j < 4; ++j) {
            float f = c[j];
            if (f >= 0.0f && f <= 1.0f) {
                int idx = (int)(f * (float)BINS);   // trunc == floor for f >= 0
                idx = idx > (BINS - 1) ? (BINS - 1) : idx;  // f == 1.0 -> last bin
                atomicAdd(&lh[wave][idx], 1u);
            }
        }
    }
    __syncthreads();

    // flush: reduce the 8 wave-private copies, one global atomic per bin
    for (int bin = tid; bin < BINS; bin += HIST_THREADS) {
        unsigned int s = 0;
        #pragma unroll
        for (int w = 0; w < WAVES_PER_BLOCK; ++w) s += lh[w][bin];
        atomicAdd(&counts[slice * BINS + bin], s);
    }
}

// ---------------------------------------------------------------------------
// Pass 2: out[64,128] = relu(feats[64,384] @ W.T + b) via v_wmma_f32_16x16x32_f16.
// One 1024-thread block = 32 waves; wave w owns output tile (mi = w/8, ni = w%8).
// feats[m][k] = counts[m*384+k] / 262144, converted to f16 on the fly.
// A-fragment (16x32 f16) per-lane layout: lane l (row = l&15):
//   e[0..7]  = K + ((l<16)?0:8) + j
//   e[8..15] = K + 16 + ((l<16)?0:8) + j
// B loads W rows with identical indexing (B[k][n] = W[n][k], symmetric layout).
// ---------------------------------------------------------------------------
__global__ __launch_bounds__(1024)
void gemm_wmma_kernel(const unsigned int* __restrict__ counts,
                      const float* __restrict__ W,
                      const float* __restrict__ bias,
                      float* __restrict__ out) {
    const int tid  = threadIdx.x;
    const int lane = tid & 31;
    const int wave = tid >> 5;
    const int mi = wave >> 3;                 // 0..3  (M tiles of 16 -> 64 rows)
    const int ni = wave & 7;                  // 0..7  (N tiles of 16 -> 128 cols)

    const int r     = lane & 15;
    const int khalf = (lane >> 4) << 3;       // 0 or 8

    const int arow = mi * 16 + r;             // feats row  (0..63)
    const int brow = ni * 16 + r;             // W row      (0..127)

    const float inv = 1.0f / (float)HW;

    v8f acc = {};
    #pragma unroll
    for (int k0 = 0; k0 < KTOT; k0 += 32) {
        const unsigned int* __restrict__ ap = counts + arow * KTOT + k0 + khalf;
        const float*        __restrict__ bp = W      + brow * KTOT + k0 + khalf;

        v16h a, bf;
        #pragma unroll
        for (int j = 0; j < 8; ++j) {
            a[j]      = (_Float16)((float)ap[j]      * inv);
            a[j + 8]  = (_Float16)((float)ap[j + 16] * inv);
            bf[j]     = (_Float16)bp[j];
            bf[j + 8] = (_Float16)bp[j + 16];
        }
        // (neg_a, A, neg_b, B, c_mod, C, reuse_a, reuse_b)
        acc = __builtin_amdgcn_wmma_f32_16x16x32_f16(
            false, a, false, bf, (short)0, acc, false, false);
    }

    // C/D layout: VGPR r2, lane l -> M = 16*mi + r2 + (l>=16 ? 8 : 0), N = 16*ni + (l&15)
    const int n    = ni * 16 + (lane & 15);
    const int mofs = (lane >> 4) << 3;
    const float bv = bias[n];
    #pragma unroll
    for (int r2 = 0; r2 < 8; ++r2) {
        const int m = mi * 16 + r2 + mofs;
        float v = acc[r2] + bv;
        out[m * NOUT + n] = v > 0.0f ? v : 0.0f;
    }
}

// ---------------------------------------------------------------------------
// Launch
// ---------------------------------------------------------------------------
extern "C" void kernel_launch(void* const* d_in, const int* in_sizes, int n_in,
                              void* d_out, int out_size, void* d_ws, size_t ws_size,
                              hipStream_t stream) {
    const float* x  = (const float*)d_in[0];   // [64, 3, 512, 512]
    const float* W  = (const float*)d_in[1];   // [128, 384]
    const float* b  = (const float*)d_in[2];   // [128]
    float* out      = (float*)d_out;           // [64, 128]

    unsigned int* counts = (unsigned int*)d_ws;  // NSLICES*BINS u32 = 96 KB
    const int ncounts = NSLICES * BINS;

    hist_zero_kernel<<<(ncounts + 255) / 256, 256, 0, stream>>>(counts, ncounts);

    hist_kernel<<<NSLICES * BLOCKS_PER_SLICE, HIST_THREADS, 0, stream>>>(x, counts);

    gemm_wmma_kernel<<<1, 1024, 0, stream>>>(counts, W, b, out);
}
